// EquAttentionPerChannel_65730179498530
// MI455X (gfx1250) — matrix operand order repaired and checked by reading, not compiled, not run on hardware
//
#include <hip/hip_runtime.h>
#include <hip/hip_bf16.h>
#include <stdint.h>

// ---------------------------------------------------------------------------
// CDNA5 / gfx1250 WMMA types
// ---------------------------------------------------------------------------
typedef __attribute__((ext_vector_type(16))) __bf16   v16bf;
typedef __attribute__((ext_vector_type(8)))  float    v8f;
typedef __attribute__((ext_vector_type(4)))  unsigned u32x4;

union AFrag {
    v16bf    v;
    u32x4    q[2];
    unsigned u[8];
};

static __device__ inline uint16_t f2bf(float f) {
    union { float f; uint32_t u; } c; c.f = f;
    uint32_t r = c.u + 0x7FFFu + ((c.u >> 16) & 1u);   // round-to-nearest-even
    return (uint16_t)(r >> 16);
}

static __device__ inline v8f wmma_bf16(const AFrag& a, const AFrag& b, v8f c) {
    return __builtin_amdgcn_wmma_f32_16x16x32_bf16(
        /*neg_a=*/false, a.v, /*neg_b=*/false, b.v,
        /*c_mod=*/(short)0, c, /*reuse_a=*/false, /*reuse_b=*/false);
}

// ---------------------------------------------------------------------------
// Row-of-16 reductions via DPP16 row rotations (pure VALU, no LDS/bpermute).
// WMMA C-layout stripes row m = r + 8*(lane>>4) across the 16 lanes of each
// half-wave, and DPP16 row ops operate on rows of 16 lanes, so rotating by
// 1,2,4,8 and combining yields a full per-row reduction in 8 VALU ops.
// ---------------------------------------------------------------------------
template <int N>
static __device__ inline float dpp_ror(float x) {
    int i = __builtin_amdgcn_update_dpp(0, __float_as_int(x),
                                        0x120 | N, 0xF, 0xF, false); // row_ror:N
    return __int_as_float(i);
}
static __device__ inline float rmax16(float v) {
    v = fmaxf(v, dpp_ror<1>(v));
    v = fmaxf(v, dpp_ror<2>(v));
    v = fmaxf(v, dpp_ror<4>(v));
    v = fmaxf(v, dpp_ror<8>(v));
    return v;
}
static __device__ inline float rsum16(float v) {
    v += dpp_ror<1>(v);
    v += dpp_ror<2>(v);
    v += dpp_ror<4>(v);
    v += dpp_ror<8>(v);
    return v;
}

// ---------------------------------------------------------------------------
// Pack block-circulant weight W = [[A,B],[B,A]] (1024x1024) into bf16.
// ---------------------------------------------------------------------------
__global__ void pack_w_kernel(const float* __restrict__ A,
                              const float* __restrict__ Bm,
                              uint16_t* __restrict__ dst) {
    int idx = blockIdx.x * 256 + threadIdx.x;      // 1<<20 total
    int r = idx >> 10, c = idx & 1023;
    float v;
    if (r < 512) v = (c < 512) ? A[r * 512 + c]          : Bm[r * 512 + (c - 512)];
    else         v = (c < 512) ? Bm[(r - 512) * 512 + c] : A[(r - 512) * 512 + (c - 512)];
    dst[idx] = f2bf(v);
}

__global__ void f32_to_bf16_kernel(const float* __restrict__ src,
                                   uint16_t* __restrict__ dst, int n) {
    int idx = blockIdx.x * 256 + threadIdx.x;
    if (idx < n) dst[idx] = f2bf(src[idx]);
}

// VT[inst][d][key]  (inst = b*32 + h*2 + g; 32 dims; 1024 keys)
__global__ void transpose_v_kernel(const uint16_t* __restrict__ qkv,
                                   uint16_t* __restrict__ vt) {
    int idx = blockIdx.x * 256 + threadIdx.x;      // 4*32*32*1024 = 4194304
    int key  = idx & 1023;
    int d    = (idx >> 10) & 31;
    int inst = idx >> 15;
    int b = inst >> 5, hg = inst & 31;
    vt[idx] = qkv[(size_t)(b * 1024 + key) * 3072 + 2048 + hg * 32 + d];
}

// ---------------------------------------------------------------------------
// bf16 WMMA GEMM:  C(MxN) = X(MxK) * W(NxK)^T ; fp32 accumulate.
// Block = 128 threads (4 waves); wave tile 32Mx64N; block tile 128Mx64N.
// A/B operand layouts (ISA 7.12.2) map to contiguous 128-bit loads.
// ---------------------------------------------------------------------------
template <int OUT_BF16>
__global__ __launch_bounds__(128)
void gemm_bf16_kernel(const uint16_t* __restrict__ X,
                      const uint16_t* __restrict__ W,
                      void* __restrict__ Cp, int M, int N, int K) {
    const int lane = threadIdx.x & 31;
    const int wave = threadIdx.x >> 5;
    const int hi   = lane >> 4;
    const int l16  = lane & 15;
    const int m0   = blockIdx.y * 128 + wave * 32;
    const int n0   = blockIdx.x * 64;
    const int Ku4  = K >> 3;                       // row length in uint4 units

    v8f acc[2][4] = {};
    const u32x4* Xq = (const u32x4*)X;
    const u32x4* Wq = (const u32x4*)W;

    for (int kt = 0; kt < K; kt += 32) {
        AFrag a0, a1;
        const int ak = (kt >> 3) + hi;             // A: dword grps [0..3]/[4..7] + [8..11]/[12..15]
        a0.q[0] = Xq[(size_t)(m0 + l16) * Ku4 + ak];
        a0.q[1] = Xq[(size_t)(m0 + l16) * Ku4 + ak + 2];
        a1.q[0] = Xq[(size_t)(m0 + 16 + l16) * Ku4 + ak];
        a1.q[1] = Xq[(size_t)(m0 + 16 + l16) * Ku4 + ak + 2];
        const int bk = (kt >> 3) + hi;             // B: dword grps [0..3]/[4..7] + [8..11]/[12..15]
        #pragma unroll
        for (int j = 0; j < 4; ++j) {
            AFrag b;
            size_t bb = (size_t)(n0 + j * 16 + l16) * Ku4 + bk;
            b.q[0] = Wq[bb]; b.q[1] = Wq[bb + 2];
            acc[0][j] = wmma_bf16(a0, b, acc[0][j]);
            acc[1][j] = wmma_bf16(a1, b, acc[1][j]);
        }
    }

    #pragma unroll
    for (int mt = 0; mt < 2; ++mt)
        #pragma unroll
        for (int j = 0; j < 4; ++j)
            #pragma unroll
            for (int r = 0; r < 8; ++r) {
                int m = m0 + mt * 16 + r + hi * 8; // C layout: M = r + 8*(lane>>4)
                int n = n0 + j * 16 + l16;         //            N = lane&15
                float val = acc[mt][j][r];
                if (OUT_BF16) ((uint16_t*)Cp)[(size_t)m * N + n] = f2bf(val);
                else          ((float*)Cp)[(size_t)m * N + n]    = val;
            }
}

// ---------------------------------------------------------------------------
// Flash attention, one (b,h,g) instance per 16 blocks; 128 instances.
// Block = 128 threads (4 waves); each wave owns 16 query rows, loops all
// 1024 keys in 64-key tiles: 4 WMMA (S=QK^T) + 4 WMMA (O += P*V) per tile.
// P converted C-layout -> A-layout through a private 4KB LDS tile per wave.
// Softmax row stats reduced with DPP16 row rotations (no ds_bpermute).
// ---------------------------------------------------------------------------
__global__ __launch_bounds__(128)
void attn_kernel(const uint16_t* __restrict__ qkv,     // (4096 x 3072) bf16: q|k|v
                 const uint16_t* __restrict__ vt,      // VT[inst][32][1024] bf16
                 uint16_t* __restrict__ outO) {        // (4096 x 1024) bf16
    __shared__ float lds[4 * 16 * 64];

    const int lane = threadIdx.x & 31;
    const int wave = threadIdx.x >> 5;
    const int hi   = lane >> 4;
    const int l16  = lane & 15;
    const int inst = blockIdx.x >> 4;                  // 0..127
    const int qblk = blockIdx.x & 15;
    const int b    = inst >> 5, hg = inst & 31;
    const int c0   = hg * 32;
    const int qrow0 = qblk * 64 + wave * 16;           // row within this batch
    const size_t rowBase = (size_t)b * 1024;

    const u32x4* Q4 = (const u32x4*)qkv;               // row = 3072 elems = 384 uint4
    const u32x4* V4 = (const u32x4*)vt;

    // Q fragment (16 rows x 32 dims), loaded once
    AFrag qa;
    {
        size_t qi = (rowBase + qrow0 + l16) * 384 + (size_t)hg * 4 + hi;
        qa.q[0] = Q4[qi]; qa.q[1] = Q4[qi + 2];
    }

    v8f o0 = {}, o1 = {};
    float mrow[8], lrow[8];
    #pragma unroll
    for (int r = 0; r < 8; ++r) { mrow[r] = -1e30f; lrow[r] = 0.0f; }

    float* myLds = lds + wave * 1024;                  // 16 x 64 f32
    const float SCLE = 0.17677669529663687f * 1.44269504088896341f; // hd^-0.5 * log2(e)

    for (int j0 = 0; j0 < 1024; j0 += 64) {
        // --- S = Q K^T for 64 keys: four 16-key B subtiles -------------------
        v8f s[4];
        #pragma unroll
        for (int js = 0; js < 4; ++js) {
            AFrag kb;
            size_t ki = (rowBase + j0 + js * 16 + l16) * 384 + 128 +
                        (size_t)hg * 4 + hi;
            kb.q[0] = Q4[ki]; kb.q[1] = Q4[ki + 2];
            v8f z = {};
            s[js] = wmma_bf16(qa, kb, z);
        }

        // --- Online softmax per row (row m = r + 8*hi over 16 lanes) ---------
        #pragma unroll
        for (int r = 0; r < 8; ++r) {
            float a0 = s[0][r] * SCLE;
            float a1 = s[1][r] * SCLE;
            float a2 = s[2][r] * SCLE;
            float a3 = s[3][r] * SCLE;
            float tm   = rmax16(fmaxf(fmaxf(a0, a1), fmaxf(a2, a3)));
            float mnew = fmaxf(mrow[r], tm);
            float corr = exp2f(mrow[r] - mnew);
            float e0 = exp2f(a0 - mnew);
            float e1 = exp2f(a1 - mnew);
            float e2 = exp2f(a2 - mnew);
            float e3 = exp2f(a3 - mnew);
            float ts = rsum16((e0 + e1) + (e2 + e3));
            lrow[r] = lrow[r] * corr + ts;
            mrow[r] = mnew;
            o0[r] *= corr;
            o1[r] *= corr;
            float* rowp = myLds + (r + hi * 8) * 64 + l16;
            rowp[0]  = e0;
            rowp[16] = e1;
            rowp[32] = e2;
            rowp[48] = e3;
        }
        asm volatile("s_wait_dscnt 0" ::: "memory");

        // --- Reload P in WMMA A-layout: two 16x32 frags (keys 0..31, 32..63) -
        AFrag pa0, pa1;
        {
            const float* mrowp = myLds + l16 * 64;
            const int kb = hi * 16;           // lanes<16: K 0..15 ; >=16: 16..31
            #pragma unroll
            for (int i = 0; i < 4; ++i) {
                int k = kb + 2 * i;
                pa0.u[i]     = (uint32_t)f2bf(mrowp[k])      | ((uint32_t)f2bf(mrowp[k + 1]) << 16);
                pa0.u[4 + i] = (uint32_t)f2bf(mrowp[8 + k])  | ((uint32_t)f2bf(mrowp[9 + k]) << 16);
                pa1.u[i]     = (uint32_t)f2bf(mrowp[32 + k]) | ((uint32_t)f2bf(mrowp[33 + k]) << 16);
                pa1.u[4 + i] = (uint32_t)f2bf(mrowp[40 + k]) | ((uint32_t)f2bf(mrowp[41 + k]) << 16);
            }
        }

        // --- O += P * V : V B-operands from transposed V (contiguous keys) ---
        size_t vbase = (size_t)inst * 32768;
        AFrag vb00, vb01, vb10, vb11;
        {
            size_t vi00 = (vbase + (size_t)(l16)      * 1024 + j0      + hi * 16) >> 3;
            size_t vi01 = (vbase + (size_t)(16 + l16) * 1024 + j0      + hi * 16) >> 3;
            size_t vi10 = (vbase + (size_t)(l16)      * 1024 + j0 + 32 + hi * 16) >> 3;
            size_t vi11 = (vbase + (size_t)(16 + l16) * 1024 + j0 + 32 + hi * 16) >> 3;
            vb00.q[0] = V4[vi00]; vb00.q[1] = V4[vi00 + 1];
            vb01.q[0] = V4[vi01]; vb01.q[1] = V4[vi01 + 1];
            vb10.q[0] = V4[vi10]; vb10.q[1] = V4[vi10 + 1];
            vb11.q[0] = V4[vi11]; vb11.q[1] = V4[vi11 + 1];
        }
        o0 = wmma_bf16(pa0, vb00, o0);   // keys 0..31,  dims 0..15
        o1 = wmma_bf16(pa0, vb01, o1);   // keys 0..31,  dims 16..31
        o0 = wmma_bf16(pa1, vb10, o0);   // keys 32..63, dims 0..15
        o1 = wmma_bf16(pa1, vb11, o1);   // keys 32..63, dims 16..31
    }

    // Normalize and write O as bf16: row = global token, col = c0 + d
    #pragma unroll
    for (int r = 0; r < 8; ++r) {
        float inv = 1.0f / lrow[r];
        size_t mglob = rowBase + qrow0 + r + hi * 8;
        int col0 = c0 + l16;
        outO[mglob * 1024 + col0]      = f2bf(o0[r] * inv);
        outO[mglob * 1024 + col0 + 16] = f2bf(o1[r] * inv);
    }
}

// ---------------------------------------------------------------------------
// Launch
// ---------------------------------------------------------------------------
extern "C" void kernel_launch(void* const* d_in, const int* in_sizes, int n_in,
                              void* d_out, int out_size, void* d_ws, size_t ws_size,
                              hipStream_t stream) {
    const float* x   = (const float*)d_in[0];
    const float* wqA = (const float*)d_in[1];
    const float* wqB = (const float*)d_in[2];
    const float* wkA = (const float*)d_in[3];
    const float* wkB = (const float*)d_in[4];
    const float* wvA = (const float*)d_in[5];
    const float* wvB = (const float*)d_in[6];
    const float* wpA = (const float*)d_in[7];
    const float* wpB = (const float*)d_in[8];

    // Workspace layout (bf16 elements):
    uint16_t* base = (uint16_t*)d_ws;
    uint16_t* Xb    = base;                 //  4096*1024          =  4194304 ( 8MB)
    uint16_t* Wqkv  = base + 4194304;       //  3072*1024          =  3145728 ( 6MB)
    uint16_t* Wp    = base + 7340032;       //  1024*1024          =  1048576 ( 2MB)
    uint16_t* QKV   = base + 8388608;       //  4096*3072          = 12582912 (24MB)
    uint16_t* VT    = base + 20971520;      //  128*32*1024        =  4194304 ( 8MB)
    uint16_t* Ob    = base + 25165824;      //  4096*1024          =  4194304 ( 8MB)

    // 1) Convert x and pack weights to bf16
    f32_to_bf16_kernel<<<16384, 256, 0, stream>>>(x, Xb, 4194304);
    pack_w_kernel<<<4096, 256, 0, stream>>>(wqA, wqB, Wqkv);
    pack_w_kernel<<<4096, 256, 0, stream>>>(wkA, wkB, Wqkv + 1048576);
    pack_w_kernel<<<4096, 256, 0, stream>>>(wvA, wvB, Wqkv + 2097152);
    pack_w_kernel<<<4096, 256, 0, stream>>>(wpA, wpB, Wp);

    // 2) Fused QKV projection: (4096x1024) x (3072x1024)^T -> bf16
    {
        dim3 grid(3072 / 64, 4096 / 128);
        gemm_bf16_kernel<1><<<grid, 128, 0, stream>>>(Xb, Wqkv, QKV, 4096, 3072, 1024);
    }

    // 3) Transpose V per (b,h,g) instance for contiguous P*V B-operand loads
    transpose_v_kernel<<<16384, 256, 0, stream>>>(QKV, VT);

    // 4) Flash attention over 128 instances x 16 query blocks
    attn_kernel<<<2048, 128, 0, stream>>>(QKV, VT, Ob);

    // 5) Output projection: (4096x1024) x (1024x1024)^T -> fp32 d_out
    {
        dim3 grid(1024 / 64, 4096 / 128);
        gemm_bf16_kernel<0><<<grid, 128, 0, stream>>>(Ob, Wp, d_out, 4096, 1024, 1024);
    }
}
